// CTM_48713519071884
// MI455X (gfx1250) — matrix operand order repaired and checked by reading, not compiled, not run on hardware
//
#include <hip/hip_runtime.h>
#include <stdint.h>

// ---------------------------------------------------------------------------
// Problem constants (from reference): T=2048, B=8, D=1024, K=2
// ---------------------------------------------------------------------------
#define T_DIM 2048
#define B_DIM 8
#define D_DIM 1024
#define M_TOT (T_DIM * B_DIM) // 16384 rows of the flattened (t,b) GEMMs

typedef __attribute__((ext_vector_type(16))) __bf16 v16bf;
typedef __attribute__((ext_vector_type(8)))  float  v8f;
typedef unsigned int u32x4 __attribute__((ext_vector_type(4)));
typedef int          i32x8 __attribute__((ext_vector_type(8)));
typedef int          i32x4 __attribute__((ext_vector_type(4)));

#if __has_builtin(__builtin_amdgcn_tensor_load_to_lds)
#define HAVE_TDM 1
#else
#define HAVE_TDM 0
#endif

// f32 -> bf16 with round-to-nearest-even
__device__ __forceinline__ unsigned short f2bf(float x) {
  unsigned int u = __float_as_uint(x);
  unsigned int r = (u + 0x7FFFu + ((u >> 16) & 1u)) >> 16;
  return (unsigned short)r;
}

// ---------------------------------------------------------------------------
// Prep 1: Apad (bf16): rows 0..B-1 = hidden_state broadcast, rows B.. = vec.
// The conv GEMM reads row r for the W0 half and row r+B for the W1 half.
// ---------------------------------------------------------------------------
__global__ void prep_apad(const float* __restrict__ vec,
                          const float* __restrict__ hidden,
                          unsigned short* __restrict__ apad) {
  size_t i = ((size_t)blockIdx.x * blockDim.x + threadIdx.x) * 4;
  const size_t hidEnd = (size_t)B_DIM * D_DIM;
  const size_t total  = (size_t)(M_TOT + B_DIM) * D_DIM;
  if (i >= total) return;
  float4 f;
  if (i < hidEnd) {
    int d = (int)(i & (D_DIM - 1)); // rows 0..7 all replicate hidden_state[0,:]
    f = *(const float4*)(hidden + d);
  } else {
    f = *(const float4*)(vec + (i - hidEnd));
  }
  ushort4 p;
  p.x = f2bf(f.x); p.y = f2bf(f.y); p.z = f2bf(f.z); p.w = f2bf(f.w);
  *(ushort4*)(apad + i) = p;
}

// ---------------------------------------------------------------------------
// Prep 2: WcT[n][kk] = bf16(conv_w[kk>=1024][kk&1023][n]); K-contiguous so the
// GEMM's B-fragment gathers from LDS are contiguous b128s.
// ---------------------------------------------------------------------------
__global__ void prep_wconv(const float* __restrict__ conv_w,
                           unsigned short* __restrict__ wct) {
  int t  = (blockIdx.x * blockDim.x + threadIdx.x) * 4; // 1024*2048 elems
  int n  = t >> 11;
  int kk = t & 2047;
  const float* src = conv_w + (size_t)(kk >> 10) * D_DIM * D_DIM
                            + (size_t)(kk & 1023) * D_DIM + n;
  ushort4 p;
  p.x = f2bf(src[0]);
  p.y = f2bf(src[D_DIM]);
  p.z = f2bf(src[2 * D_DIM]);
  p.w = f2bf(src[3 * D_DIM]);
  *(ushort4*)(wct + (size_t)n * 2048 + kk) = p;
}

// Prep 3: WoT[n][k] = bf16(olin_w[k][n])
__global__ void prep_wolin(const float* __restrict__ w,
                           unsigned short* __restrict__ wt) {
  int t = (blockIdx.x * blockDim.x + threadIdx.x) * 4; // 1024*1024 elems
  int n = t >> 10;
  int k = t & 1023;
  const float* src = w + (size_t)k * D_DIM + n;
  ushort4 p;
  p.x = f2bf(src[0]);
  p.y = f2bf(src[D_DIM]);
  p.z = f2bf(src[2 * D_DIM]);
  p.w = f2bf(src[3 * D_DIM]);
  *(ushort4*)(wt + (size_t)n * 1024 + k) = p;
}

#if HAVE_TDM
// ---------------------------------------------------------------------------
// TDM: DMA a 2-D bf16 tile (tile_d0 elems wide x tile_d1 rows) from global to
// LDS, with pad_amount=4dw per pad_interval=16dw -> 64B rows padded to 80B in
// LDS (matches the [128][40]-ushort tile layout the fragment gathers use).
// D# bitfields per CDNA5 ISA ch.8 (group0 128b, group1 256b; groups 2/3 and
// the trailing group are zero for <=2D tensors). 6-arg builtin form
// (clang-23 / therock-10.0 lane).
// ---------------------------------------------------------------------------
__device__ __forceinline__ void tdm_load_tile_2d(uint64_t gaddr, uint32_t lds_off,
                                                 uint32_t stride_elems,
                                                 uint32_t tile_d0, uint32_t tile_d1) {
  u32x4 g0;
  g0[0] = 1u;                                           // count=1, user D#
  g0[1] = lds_off;                                      // lds_addr [63:32]
  g0[2] = (uint32_t)gaddr;                              // global_addr lo
  g0[3] = ((uint32_t)(gaddr >> 32) & 0x01FFFFFFu)       // global_addr [56:32]
          | (2u << 30);                                 // type=2 ("image")
  i32x8 g1;
  g1[0] = (int)((1u << 16)                              // data_size: 2 bytes
              | (1u << 20)                              // pad_enable
              | (3u << 22)                              // pad_interval: 16 dw
              | (3u << 25));                            // pad_amount: 4 dw
  g1[1] = (int)((tile_d0 & 0xFFFFu) << 16);             // tensor_dim0 lo16
  g1[2] = (int)((tile_d0 >> 16) | ((tile_d1 & 0xFFFFu) << 16)); // d0 hi, d1 lo
  g1[3] = (int)((tile_d1 >> 16) | ((tile_d0 & 0xFFFFu) << 16)); // d1 hi, tile_dim0
  g1[4] = (int)(tile_d1 & 0xFFFFu);                     // tile_dim1 (tile_dim2=0)
  g1[5] = (int)stride_elems;                            // tensor_dim0_stride lo
  g1[6] = 0;                                            // stride hi, dim1_stride lo
  g1[7] = 0;
  i32x4 z4 = {0, 0, 0, 0};
  i32x8 z8 = {0, 0, 0, 0, 0, 0, 0, 0};
  __builtin_amdgcn_tensor_load_to_lds(g0, g1, z4, z4, z8, 0);
}
#endif

// ---------------------------------------------------------------------------
// bf16 WMMA GEMM:  C[M=16384][1024] = A @ Bt^T + bias
//   A  : bf16 row-major, row stride 1024; element addr = r*1024 + k
//        + (k >= kSwitch ? kExtra : 0)   (kExtra=7168 == +B rows for the
//        second half of the conv K dimension)
//   Bt : bf16 [N=1024][Ktot], K-contiguous (pre-transposed weights)
// Block: 256 threads = 8 waves; block tile 128Mx128N; wave tile 32Mx64N.
// Double-buffered LDS fed by the Tensor Data Mover (wave 0 issues the DMA;
// the s_wait_tensorcnt 0 is executed by ALL waves — it is an immediate no-op
// for waves with TENSORcnt==0, and keeping it unconditional avoids the
// saveexec/branch around the wait), so the DMA of tile i+1 overlaps the
// 8 WMMAs of tile i.
// ---------------------------------------------------------------------------
__global__ __launch_bounds__(256)
void gemm_bf16(const unsigned short* __restrict__ A,
               const unsigned short* __restrict__ Bt,
               const float* __restrict__ bias,
               float* __restrict__ C,
               int Ktot, int kSwitch, int kExtra) {
  __shared__ __align__(16) unsigned short lA[2][128][40];
  __shared__ __align__(16) unsigned short lB[2][128][40];

  const int tid   = threadIdx.x;
  const int lane  = tid & 31;
  const int wave  = tid >> 5;
  const int waveM = wave & 3;  // 0..3 -> 32-row sub-tile
  const int waveN = wave >> 2; // 0..1 -> 64-col sub-tile

  const int mBlk = blockIdx.y * 128;
  const int nBlk = blockIdx.x * 128;

  v8f acc0[4] = {}; // rows waveM*32 + 0..15, 4 x 16 cols
  v8f acc1[4] = {}; // rows waveM*32 + 16..31

  // Per-lane fragment geometry (ISA 16-bit A 16x32 / B 32x16 layouts, wave32)
  const int khA   = (lane >> 4) * 8;  // A: K-half 0 or 8
  const int khB   = (lane >> 4) * 8;  // B: K-half 0 or 8
  const int arow0 = waveM * 32 + (lane & 15);
  const int arow1 = arow0 + 16;

#if !HAVE_TDM
  // Fallback fill split: 2 chunks of 8 ushorts per thread per tile
  const int aRow0 = tid >> 2, aCol0 = (tid & 3) * 8;          // chunk tid
  const int aRow1 = (tid + 256) >> 2, aCol1 = ((tid + 256) & 3) * 8;
#endif

  int stage = 0;

#if HAVE_TDM
  // Prologue: DMA tile k0=0 into stage 0
  if (wave == 0) {
    const int kA0 = (0 >= kSwitch) ? kExtra : 0;
    tdm_load_tile_2d((uint64_t)(A + (size_t)mBlk * 1024 + kA0),
                     (uint32_t)(uintptr_t)&lA[0][0][0], 1024u, 32u, 128u);
    tdm_load_tile_2d((uint64_t)(Bt + (size_t)nBlk * (size_t)Ktot),
                     (uint32_t)(uintptr_t)&lB[0][0][0], (uint32_t)Ktot, 32u, 128u);
  }
  __builtin_amdgcn_s_wait_tensorcnt(0); // no-op for waves that issued nothing
  __syncthreads();
#endif

  for (int k0 = 0; k0 < Ktot; k0 += 32) {
#if HAVE_TDM
    // Kick off the DMA of tile k0+32 into the other stage; it overlaps the
    // WMMAs below and is closed by the tensorcnt wait + barrier at loop end.
    if (wave == 0 && (k0 + 32) < Ktot) {
      const int kn = k0 + 32;
      const int kAn = kn + ((kn >= kSwitch) ? kExtra : 0);
      tdm_load_tile_2d((uint64_t)(A + (size_t)mBlk * 1024 + kAn),
                       (uint32_t)(uintptr_t)&lA[stage ^ 1][0][0], 1024u, 32u, 128u);
      tdm_load_tile_2d((uint64_t)(Bt + (size_t)nBlk * (size_t)Ktot + kn),
                       (uint32_t)(uintptr_t)&lB[stage ^ 1][0][0], (uint32_t)Ktot, 32u, 128u);
    }
#else
    {
      const int kA = k0 + ((k0 >= kSwitch) ? kExtra : 0);
      const uint4 a0 = *(const uint4*)(A  + (size_t)(mBlk + aRow0) * 1024 + kA + aCol0);
      const uint4 a1 = *(const uint4*)(A  + (size_t)(mBlk + aRow1) * 1024 + kA + aCol1);
      const uint4 b0 = *(const uint4*)(Bt + (size_t)(nBlk + aRow0) * Ktot + k0 + aCol0);
      const uint4 b1 = *(const uint4*)(Bt + (size_t)(nBlk + aRow1) * Ktot + k0 + aCol1);
      __syncthreads();
      *(uint4*)(&lA[0][aRow0][aCol0]) = a0;
      *(uint4*)(&lA[0][aRow1][aCol1]) = a1;
      *(uint4*)(&lB[0][aRow0][aCol0]) = b0;
      *(uint4*)(&lB[0][aRow1][aCol1]) = b1;
      __syncthreads();
    }
#endif

    const unsigned short (*tA)[40] = lA[stage];
    const unsigned short (*tB)[40] = lB[stage];

    // A fragments: lane holds row, K in {khA..khA+7} U {16+khA..16+khA+7}
    union Frag { uint4 q[2]; v16bf v; };
    Frag ua0, ua1;
    ua0.q[0] = *(const uint4*)(&tA[arow0][khA]);
    ua0.q[1] = *(const uint4*)(&tA[arow0][16 + khA]);
    ua1.q[0] = *(const uint4*)(&tA[arow1][khA]);
    ua1.q[1] = *(const uint4*)(&tA[arow1][16 + khA]);

#pragma unroll
    for (int s = 0; s < 4; ++s) {
      Frag ub; // lane holds col, K in {khB..khB+7} U {16+khB..16+khB+7}
      const unsigned short* bp = &tB[waveN * 64 + s * 16 + (lane & 15)][khB];
      ub.q[0] = *(const uint4*)(bp);
      ub.q[1] = *(const uint4*)(bp + 16);
      acc0[s] = __builtin_amdgcn_wmma_f32_16x16x32_bf16(
          false, ua0.v, false, ub.v, (short)0, acc0[s], false, false);
      acc1[s] = __builtin_amdgcn_wmma_f32_16x16x32_bf16(
          false, ua1.v, false, ub.v, (short)0, acc1[s], false, false);
    }

#if HAVE_TDM
    __builtin_amdgcn_s_wait_tensorcnt(0); // wave0: close DMA; others: no-op
    __syncthreads();
    stage ^= 1;
#endif
  }

  // C layout: lanes 0-15 -> N=lane, M=g ; lanes 16-31 -> N=lane-16, M=8+g
  const int nBase = nBlk + waveN * 64 + (lane & 15);
  const int mBase = mBlk + waveM * 32 + ((lane >> 4) * 8);
#pragma unroll
  for (int s = 0; s < 4; ++s) {
    const int col = nBase + s * 16;
    const float bb = bias[col];
#pragma unroll
    for (int g = 0; g < 8; ++g) {
      C[(size_t)(mBase + g) * 1024 + col]      = acc0[s][g] + bb;
      C[(size_t)(mBase + 16 + g) * 1024 + col] = acc1[s][g] + bb;
    }
  }
}

// ---------------------------------------------------------------------------
// Sequential scan: one thread per (b,d) lane, T steps in f32.
//   h = c*lv + cv ; c' = h*cv
// ---------------------------------------------------------------------------
__global__ __launch_bounds__(256)
void scan_kernel(const float* __restrict__ vec,
                 const float* __restrict__ cv,
                 const float* __restrict__ cell0,
                 float* __restrict__ hs_out,
                 float* __restrict__ cfin_out,
                 unsigned short* __restrict__ hs_bf) {
  const int u = blockIdx.x * blockDim.x + threadIdx.x; // 0..B*D-1
  const int d = u & (D_DIM - 1);
  float c = cell0[d];
  size_t idx = (size_t)u;
#pragma unroll 4
  for (int t = 0; t < T_DIM; ++t, idx += (size_t)B_DIM * D_DIM) {
    const float lv  = vec[idx];
    const float cvv = cv[idx];
    const float h   = fmaf(c, lv, cvv);
    c = h * cvv;
    hs_out[idx] = h;
    hs_bf[idx]  = f2bf(h);
  }
  cfin_out[u] = c;
}

// ---------------------------------------------------------------------------
// Launch: prep casts -> conv GEMM -> scan -> output GEMM
// Workspace layout (bytes):
//   [0)        cv (f32, 16384x1024)            67,108,864 B
//   [..]       Apad bf16 (16392x1024)          33,570,816 B
//   [..]       WcT bf16 (1024x2048)             4,194,304 B
//   [..]       WoT bf16 (1024x1024)             2,097,152 B
//   [..]       hs_bf16 (16384x1024)            33,554,432 B   (~134 MB total)
// ---------------------------------------------------------------------------
extern "C" void kernel_launch(void* const* d_in, const int* in_sizes, int n_in,
                              void* d_out, int out_size, void* d_ws, size_t ws_size,
                              hipStream_t stream) {
  const float* vec   = (const float*)d_in[0];
  const float* hid   = (const float*)d_in[1];
  const float* cell  = (const float*)d_in[2];
  const float* convw = (const float*)d_in[3];
  const float* convb = (const float*)d_in[4];
  const float* olinw = (const float*)d_in[5];
  const float* olinb = (const float*)d_in[6];

  char* ws = (char*)d_ws;
  float*          cvb  = (float*)ws;
  unsigned short* apad = (unsigned short*)(ws + 67108864ull);
  unsigned short* wct  = (unsigned short*)(ws + 67108864ull + 33570816ull);
  unsigned short* wot  = (unsigned short*)(ws + 67108864ull + 33570816ull + 4194304ull);
  unsigned short* hsbf = (unsigned short*)(ws + 67108864ull + 33570816ull + 4194304ull + 2097152ull);

  float* ret  = (float*)d_out;
  float* hs   = ret + (size_t)M_TOT * D_DIM;
  float* cfin = hs  + (size_t)M_TOT * D_DIM;

  // --- prep / bf16 casts ---
  {
    size_t nThr = (size_t)(M_TOT + B_DIM) * D_DIM / 4; // 4,196,352
    prep_apad<<<(unsigned)((nThr + 255) / 256), 256, 0, stream>>>(vec, hid, apad);
  }
  prep_wconv<<<(1024 * 2048 / 4) / 256, 256, 0, stream>>>(convw, wct);
  prep_wolin<<<(1024 * 1024 / 4) / 256, 256, 0, stream>>>(olinw, wot);

  // --- conv as GEMM: M=16384, K=2048 (A shifted +B rows for k>=1024), N=1024
  dim3 gg(1024 / 128, M_TOT / 128); // (8, 128)
  gemm_bf16<<<gg, 256, 0, stream>>>(apad, wct, convb, cvb,
                                    /*Ktot=*/2048, /*kSwitch=*/1024, /*kExtra=*/7168);

  // --- recurrence ---
  scan_kernel<<<(B_DIM * D_DIM) / 256, 256, 0, stream>>>(vec, cvb, cell, hs, cfin, hsbf);

  // --- output linear: M=16384, K=1024, N=1024 ---
  gemm_bf16<<<gg, 256, 0, stream>>>(hsbf, wot, olinb, ret,
                                    /*Ktot=*/1024, /*kSwitch=*/1 << 30, /*kExtra=*/0);
}
